// ResidualVectorQuantizer_41205916238341
// MI455X (gfx1250) — matrix operand order repaired and checked by reading, not compiled, not run on hardware
//
#include <hip/hip_runtime.h>
#include <hip/hip_bf16.h>
#include <float.h>

// ---- problem constants ----
#define BB 8
#define NN 2048
#define DD 256
#define NUMQ 8
#define KK 2048
#define MTOK (BB * NN)          // 16384 tokens
#define COMMIT 0.25f
#define NSPLIT 4                // codebook splits across gridDim.y
#define KS (KK / NSPLIT)        // 512 codes per split -> 32 n-tiles
#define NTILES (KS / 16)        // 32
#define NFRAG (NTILES * 8)      // 256 B-fragments per split

typedef __attribute__((ext_vector_type(16))) __bf16 v16bf;
typedef __attribute__((ext_vector_type(8)))  __bf16 v8bf;
typedef __attribute__((ext_vector_type(4)))  __bf16 v4bf;
typedef __attribute__((ext_vector_type(8)))  float  v8f;

static __device__ __forceinline__ v16bf pack16(const __bf16* plo, const __bf16* phi) {
    v8bf lo = *(const v8bf*)plo;
    v8bf hi = *(const v8bf*)phi;
    v16bf r;
#pragma unroll
    for (int i = 0; i < 8; ++i) { r[i] = lo[i]; r[i + 8] = hi[i]; }
    return r;
}

// ---------------------------------------------------------------------------
// K0: zero output + loss accumulator, residual = x (f32 and bf16 mirror)
// ---------------------------------------------------------------------------
__global__ void rvq_init_kernel(const float* __restrict__ x,
                                float* __restrict__ res_f32,
                                __bf16* __restrict__ res_bf16,
                                float* __restrict__ out,
                                float* __restrict__ loss_acc) {
    int gid = blockIdx.x * blockDim.x + threadIdx.x;
    if (gid < MTOK * DD) {
        float v = x[gid];
        res_f32[gid]  = v;
        res_bf16[gid] = (__bf16)v;
        out[gid]      = 0.0f;
    }
    if (gid == 0) { loss_acc[0] = 0.0f; out[MTOK * DD] = 0.0f; }
}

// ---------------------------------------------------------------------------
// K1a: transposed bf16 codebooks  CT[q][d][k]  (k contiguous -> WMMA B layout)
// ---------------------------------------------------------------------------
__global__ void rvq_prep_ct_kernel(const float* __restrict__ cb,
                                   __bf16* __restrict__ ct) {
    int gid = blockIdx.x * blockDim.x + threadIdx.x;  // over NUMQ*KK*DD
    if (gid >= NUMQ * KK * DD) return;
    int q   = gid / (KK * DD);
    int rem = gid - q * (KK * DD);
    int n   = rem / DD;
    int d   = rem - n * DD;
    ct[(size_t)q * DD * KK + (size_t)d * KK + n] = (__bf16)cb[gid];
}

// ---------------------------------------------------------------------------
// K1b: c2[q][k] = ||codebook row||^2
// ---------------------------------------------------------------------------
__global__ void rvq_prep_c2_kernel(const float* __restrict__ cb,
                                   float* __restrict__ c2) {
    int gid = blockIdx.x * blockDim.x + threadIdx.x;  // over NUMQ*KK
    if (gid >= NUMQ * KK) return;
    const float4* p = (const float4*)(cb + (size_t)gid * DD);
    float s = 0.0f;
#pragma unroll 8
    for (int i = 0; i < DD / 4; ++i) {
        float4 v = p[i];
        s += v.x * v.x + v.y * v.y + v.z * v.z + v.w * v.w;
    }
    c2[gid] = s;
}

// ---------------------------------------------------------------------------
// K2: fused distance-GEMM + partial row argmin over this block's code split.
//     8 waves/block, 32 tokens/wave (two A tiles share every B fragment).
//     B loads software-pipelined depth 2.
//     dist(m,n) = c2[n] - 2 * dot(r_m, c_n)   (||r||^2 dropped: row-constant)
// ---------------------------------------------------------------------------
__global__ void __launch_bounds__(256)
rvq_argmin_kernel(const __bf16* __restrict__ res_bf16,
                  const __bf16* __restrict__ ct,
                  const float*  __restrict__ c2,
                  float* __restrict__ pdist,   // [NSPLIT][MTOK]
                  int*   __restrict__ pidx,    // [NSPLIT][MTOK]
                  int q) {
    const int tid   = threadIdx.x;
    const int wave  = tid >> 5;
    const int lane  = tid & 31;
    const int m0    = blockIdx.x * 256 + wave * 32;     // wave's 32 tokens
    const int split = blockIdx.y;
    const int nbase = split * KS;                       // first code of split
    const int kbase = (lane >> 4) * 8;                  // A-layout half select

    // Two A tile-sets: rows m0..m0+15 (lo) and m0+16..m0+31 (hi)
    v16bf aL[DD / 32], aH[DD / 32];
    {
        const __bf16* rpL = res_bf16 + (size_t)(m0 + (lane & 15)) * DD;
        const __bf16* rpH = rpL + (size_t)16 * DD;
#pragma unroll
        for (int c = 0; c < DD / 32; ++c) {
            aL[c] = pack16(rpL + c * 32 + kbase, rpL + c * 32 + 16 + kbase);
            aH[c] = pack16(rpH + c * 32 + kbase, rpH + c * 32 + 16 + kbase);
        }
    }

    const __bf16* ctq = ct + (size_t)q * DD * KK + nbase;
    const float*  c2q = c2 + (size_t)q * KK + nbase;

    // B fragment loader: fragment f = tile*8 + chunk, wraps at NFRAG
    auto loadB = [&](int f) -> v16bf {
        f &= (NFRAG - 1);
        int tile = f >> 3, c = f & 7;
        const __bf16* bp = ctq + (size_t)(c * 32 + kbase) * KK + tile * 16 + (lane & 15);
        v16bf r;
#pragma unroll
        for (int i = 0; i < 8; ++i) { r[i] = bp[(size_t)i * KK]; r[i + 8] = bp[(size_t)(i + 16) * KK]; }
        return r;
    };

    float best[16];
    int   bestn[16];
#pragma unroll
    for (int j = 0; j < 16; ++j) { best[j] = FLT_MAX; bestn[j] = 0; }

    v16bf b0 = loadB(0);
    v16bf b1 = loadB(1);

    for (int t = 0; t < NTILES; ++t) {
        v8f accL = {}, accH = {};
#pragma unroll
        for (int c = 0; c < DD / 32; ++c) {
            v16bf bn = loadB(t * 8 + c + 2);     // prefetch depth 2
            accL = __builtin_amdgcn_wmma_f32_16x16x32_bf16(
                false, aL[c], false, b0, (short)0, accL, false, false);
            accH = __builtin_amdgcn_wmma_f32_16x16x32_bf16(
                false, aH[c], false, b0, (short)0, accH, false, false);
            b0 = b1; b1 = bn;
        }
        int   n   = nbase + t * 16 + (lane & 15);
        float c2v = c2q[t * 16 + (lane & 15)];
#pragma unroll
        for (int j = 0; j < 8; ++j) {
            float dL = fmaf(-2.0f, accL[j], c2v);
            if (dL < best[j])     { best[j] = dL;     bestn[j] = n; }
            float dH = fmaf(-2.0f, accH[j], c2v);
            if (dH < best[8 + j]) { best[8 + j] = dH; bestn[8 + j] = n; }
        }
    }

    // reduce across the 16 lanes holding the same row group (wave32)
#pragma unroll
    for (int off = 8; off >= 1; off >>= 1) {
#pragma unroll
        for (int j = 0; j < 16; ++j) {
            float ob = __shfl_xor(best[j],  off, 32);
            int   on = __shfl_xor(bestn[j], off, 32);
            if (ob < best[j] || (ob == best[j] && on < bestn[j])) {
                best[j] = ob; bestn[j] = on;
            }
        }
    }

    if ((lane & 15) == 0) {
        // lane 0: rows 0..7 of each tile; lane 16: rows 8..15
        int rbase = m0 + (lane >> 4) * 8;
        float* pd = pdist + (size_t)split * MTOK;
        int*   pi = pidx  + (size_t)split * MTOK;
#pragma unroll
        for (int j = 0; j < 8; ++j) {
            pd[rbase + j]      = best[j];      pi[rbase + j]      = bestn[j];
            pd[rbase + 16 + j] = best[8 + j];  pi[rbase + 16 + j] = bestn[8 + j];
        }
    }
}

// ---------------------------------------------------------------------------
// K2b: merge the NSPLIT partial argmins (splits ascending in n; strict <
//      keeps the first occurrence, matching jnp.argmin semantics)
// ---------------------------------------------------------------------------
__global__ void rvq_argmin_merge_kernel(const float* __restrict__ pdist,
                                        const int* __restrict__ pidx,
                                        int* __restrict__ idx) {
    int t = blockIdx.x * blockDim.x + threadIdx.x;
    if (t >= MTOK) return;
    float bd = pdist[t];
    int   bn = pidx[t];
#pragma unroll
    for (int s = 1; s < NSPLIT; ++s) {
        float d = pdist[(size_t)s * MTOK + t];
        if (d < bd) { bd = d; bn = pidx[(size_t)s * MTOK + t]; }
    }
    idx[t] = bn;
}

// ---------------------------------------------------------------------------
// K3: gather q, out += q, residual -= q (f32 + bf16), accumulate sum((r-q)^2)
// ---------------------------------------------------------------------------
__global__ void __launch_bounds__(256)
rvq_update_kernel(float* __restrict__ res_f32,
                  __bf16* __restrict__ res_bf16,
                  const float* __restrict__ cb,
                  const int* __restrict__ idx,
                  float* __restrict__ out,
                  float* __restrict__ loss_acc, int q) {
    __shared__ float sred[256];
    int gid = blockIdx.x * blockDim.x + threadIdx.x;   // over MTOK*DD/4
    int t = gid >> 6;                                   // token (64 float4 each)
    int c = gid & 63;                                   // float4 chunk in D
    int n = idx[t];

    const float4* qp = (const float4*)(cb + ((size_t)q * KK + n) * DD) + c;
    float4* rp = (float4*)(res_f32 + (size_t)t * DD) + c;
    float4* op = (float4*)(out + (size_t)t * DD) + c;

    float4 qv = *qp;
    float4 rv = *rp;
    float4 ov = *op;
    float4 dv = make_float4(rv.x - qv.x, rv.y - qv.y, rv.z - qv.z, rv.w - qv.w);

    *op = make_float4(ov.x + qv.x, ov.y + qv.y, ov.z + qv.z, ov.w + qv.w);
    *rp = dv;
    v4bf db; db[0] = (__bf16)dv.x; db[1] = (__bf16)dv.y;
             db[2] = (__bf16)dv.z; db[3] = (__bf16)dv.w;
    *((v4bf*)(res_bf16 + (size_t)t * DD) + c) = db;

    float s = dv.x * dv.x + dv.y * dv.y + dv.z * dv.z + dv.w * dv.w;
    sred[threadIdx.x] = s;
    __syncthreads();
#pragma unroll
    for (int off = 128; off >= 1; off >>= 1) {
        if (threadIdx.x < off) sred[threadIdx.x] += sred[threadIdx.x + off];
        __syncthreads();
    }
    if (threadIdx.x == 0) atomicAdd(loss_acc, sred[0]);
}

// ---------------------------------------------------------------------------
// K4: total_loss = (1 + COMMIT) * sum_sq / (B*N*D)   [same mean denom/stage]
// ---------------------------------------------------------------------------
__global__ void rvq_finalize_kernel(const float* __restrict__ loss_acc,
                                    float* __restrict__ out) {
    if (threadIdx.x == 0 && blockIdx.x == 0)
        out[MTOK * DD] = (1.0f + COMMIT) * loss_acc[0] / (float)(MTOK * DD);
}

// ---------------------------------------------------------------------------
extern "C" void kernel_launch(void* const* d_in, const int* in_sizes, int n_in,
                              void* d_out, int out_size, void* d_ws, size_t ws_size,
                              hipStream_t stream) {
    const float* x  = (const float*)d_in[0];   // [B,N,D]
    const float* cb = (const float*)d_in[1];   // [NUMQ,K,D]
    float* out = (float*)d_out;                // [B*N*D] quantized_sum, then loss

    // workspace carve-out (all 256B aligned)
    char* ws = (char*)d_ws;
    size_t off = 0;
    auto alloc = [&](size_t bytes) { char* p = ws + off; off = (off + bytes + 255) & ~(size_t)255; return p; };
    float*  res_f32  = (float*) alloc((size_t)MTOK * DD * 4);
    __bf16* res_bf16 = (__bf16*)alloc((size_t)MTOK * DD * 2);
    __bf16* ct       = (__bf16*)alloc((size_t)NUMQ * DD * KK * 2);
    float*  c2       = (float*) alloc((size_t)NUMQ * KK * 4);
    int*    idx      = (int*)   alloc((size_t)MTOK * 4);
    float*  pdist    = (float*) alloc((size_t)NSPLIT * MTOK * 4);
    int*    pidx     = (int*)   alloc((size_t)NSPLIT * MTOK * 4);
    float*  loss_acc = (float*) alloc(256);

    const int elems = MTOK * DD;               // 4,194,304

    rvq_init_kernel<<<(elems + 255) / 256, 256, 0, stream>>>(x, res_f32, res_bf16, out, loss_acc);
    rvq_prep_ct_kernel<<<(NUMQ * KK * DD + 255) / 256, 256, 0, stream>>>(cb, ct);
    rvq_prep_c2_kernel<<<(NUMQ * KK + 255) / 256, 256, 0, stream>>>(cb, c2);

    dim3 agrid(MTOK / 256, NSPLIT);
    for (int q = 0; q < NUMQ; ++q) {
        rvq_argmin_kernel<<<agrid, 256, 0, stream>>>(res_bf16, ct, c2, pdist, pidx, q);
        rvq_argmin_merge_kernel<<<MTOK / 256, 256, 0, stream>>>(pdist, pidx, idx);
        rvq_update_kernel<<<elems / 4 / 256, 256, 0, stream>>>(res_f32, res_bf16, cb, idx, out, loss_acc, q);
    }
    rvq_finalize_kernel<<<1, 64, 0, stream>>>(loss_acc, out);
}
